// MultiHeadAttention_7894149890357
// MI455X (gfx1250) — matrix operand order repaired and checked by reading, not compiled, not run on hardware
//
#include <hip/hip_runtime.h>
#include <hip/hip_bf16.h>

// ---------------- problem constants ----------------
#define BB   1024
#define TT   64
#define DD   512
#define NHH  8
#define HSS  64
#define NBP_K 240   // rel_k padded rows (N-dim of P, mult of 16)
#define NBP_V 256   // rel_v padded cols (K-dim of W@RV, mult of 32)

typedef __bf16 bf16;
typedef __attribute__((ext_vector_type(16))) __bf16 v16bf;
typedef __attribute__((ext_vector_type(4)))  __bf16 v4bf;
typedef __attribute__((ext_vector_type(8)))  float  v8f;

// Scheduling pins (composable-kernel style). Masks: 0x008 = MFMA/WMMA, 0x100 = DS read.
#if defined(__has_builtin)
#if __has_builtin(__builtin_amdgcn_sched_group_barrier)
#define SGB(mask, size) __builtin_amdgcn_sched_group_barrier(mask, size, 0)
#endif
#endif
#ifndef SGB
#define SGB(mask, size)
#endif

// ---------------- WMMA fragment helpers (CDNA5 wave32 layouts) ----------------
// A 16x32 bf16: lanes 0-15 -> M=lane, K in {0..7, 16..23}; lanes 16-31 -> M=lane-16, K in {8..15, 24..31}
__device__ __forceinline__ v16bf load_a_bf(const bf16* t, int ld, int mo, int ko, int lane) {
  const int m  = mo + (lane & 15);
  const int kb = (lane & 16) ? 8 : 0;
  const bf16* row = t + m * ld + ko;
  v16bf a;
#pragma unroll
  for (int p = 0; p < 8; ++p) {
    int k = ((p & 4) << 2) + kb + 2 * (p & 3);  // p>=4 adds 16
    a[2 * p]     = row[k];
    a[2 * p + 1] = row[k + 1];
  }
  return a;
}
// same, but source is f32 in LDS (converted on the fly)
__device__ __forceinline__ v16bf load_a_f32(const float* t, int ld, int mo, int ko, int lane) {
  const int m  = mo + (lane & 15);
  const int kb = (lane & 16) ? 8 : 0;
  const float* row = t + m * ld + ko;
  v16bf a;
#pragma unroll
  for (int p = 0; p < 8; ++p) {
    int k = ((p & 4) << 2) + kb + 2 * (p & 3);
    a[2 * p]     = (bf16)row[k];
    a[2 * p + 1] = (bf16)row[k + 1];
  }
  return a;
}
// B 32x16 bf16: lanes 0-15 -> N=lane, K in {0..7, 16..23}; lanes 16-31 -> N=lane-16, K in {8..15, 24..31}
// source stored "transposed": B[k][n] = src[(no+n)*ld + ko + k]  (two runs of 8 bf16 per lane,
// works for both LDS and global (L2-resident tables) source pointers)
__device__ __forceinline__ v16bf load_bT_bf(const bf16* t, int ld, int no, int ko, int lane) {
  const int n  = no + (lane & 15);
  const int kb = ko + ((lane & 16) ? 8 : 0);
  const bf16* col = t + n * ld + kb;
  v16bf b;
#pragma unroll
  for (int q = 0; q < 16; ++q) b[q] = col[q + ((q & 8) ? 8 : 0)];
  return b;
}
__device__ __forceinline__ v8f wmma_bf(v16bf a, v16bf b, v8f c) {
  return __builtin_amdgcn_wmma_f32_16x16x32_bf16(false, a, false, b, (short)0, c, false, false);
}
__device__ __forceinline__ v8f zero8() {
  v8f c = {0.f, 0.f, 0.f, 0.f, 0.f, 0.f, 0.f, 0.f};
  return c;
}

// ---------------- kernel 0: weight / rel-table conversion to bf16 ----------------
#define N_WQKV (3 * DD * DD)   // 786432
#define N_WPRJ (DD * DD)       // 262144
#define N_RELK (NBP_K * HSS)   // 15360
#define N_RELV (HSS * NBP_V)   // 16384  (stored TRANSPOSED: (d, bucket))
#define N_CONV (N_WQKV + N_WPRJ + N_RELK + N_RELV)  // 1,080,320 = 256 * 4220

__global__ __launch_bounds__(256)
void convert_kernel(const float* __restrict__ w_qkv, const float* __restrict__ w_proj,
                    const float* __restrict__ rel_k, const float* __restrict__ rel_v,
                    bf16* __restrict__ wqkvb, bf16* __restrict__ wprojb,
                    bf16* __restrict__ relkp, bf16* __restrict__ relvT) {
  int idx = blockIdx.x * 256 + threadIdx.x;
  if (idx < N_WQKV) {
    wqkvb[idx] = (bf16)w_qkv[idx];
  } else if (idx < N_WQKV + N_WPRJ) {
    int i = idx - N_WQKV;
    wprojb[i] = (bf16)w_proj[i];
  } else if (idx < N_WQKV + N_WPRJ + N_RELK) {
    int i = idx - (N_WQKV + N_WPRJ);
    int r = i >> 6;                         // bucket row
    relkp[i] = (bf16)(r < 225 ? rel_k[i] : 0.f);
  } else if (idx < N_CONV) {
    int i  = idx - (N_WQKV + N_WPRJ + N_RELK);
    int d  = i >> 8;                        // head-dim row of transposed table
    int nb = i & 255;                       // bucket column
    relvT[i] = (bf16)(nb < 225 ? rel_v[nb * HSS + d] : 0.f);
  }
}

// ---------------- kernel 1: fused per-(batch,head) QKV + relative attention ----------------
// LDS map (bytes), reused across phases (total 145,408 < 160KB -> 2 blocks/WGP):
//   [0,8192)        Qs  bf16 64x64        (persistent)
//   [8192,16384)    Ks  bf16 64x64        (persistent)
//   [16384,24576)   Vt  bf16 64x64  (d,j) (persistent, transposed)
//   [24576,40960)   Ss  f32  64x64        (logits / exp values)
//   [40960,106496)  Ws  f32  64x256       (bucket weights)  | phase1: XA(4K)+WT(12K) | phase2-3: RKs(30K)
//   [106496,137216) Psb bf16 64x240       (rel logits, bf16 storage, f32 WMMA accum)
//   [137216,145408) As  bf16 64x64        (softmaxed attn)
#define SMEM_BYTES 145408

__global__ __launch_bounds__(128, 2)
void attn_fused(const float* __restrict__ x,
                const bf16* __restrict__ wqkv,   // (1536,512) bf16
                const bf16* __restrict__ relk,   // (240,64)   bf16 padded, (bucket, d)
                const bf16* __restrict__ relvT,  // (64,256)   bf16 padded, (d, bucket), L2-resident
                bf16* __restrict__ attn_out) {   // (B,T,D)    bf16
  extern __shared__ char smem[];
  bf16*  Qs  = (bf16*)(smem + 0);
  bf16*  Ks  = (bf16*)(smem + 8192);
  bf16*  Vt  = (bf16*)(smem + 16384);
  float* Ss  = (float*)(smem + 24576);
  float* Ws  = (float*)(smem + 40960);
  bf16*  XA  = (bf16*)(smem + 40960);
  bf16*  WT  = (bf16*)(smem + 45056);
  bf16*  RKs = (bf16*)(smem + 40960);
  bf16*  Psb = (bf16*)(smem + 106496);
  bf16*  As  = (bf16*)(smem + 137216);

  const int tid  = threadIdx.x;
  const int wave = tid >> 5;
  const int lane = tid & 31;
  const int b    = blockIdx.x >> 3;   // NH == 8
  const int h    = blockIdx.x & 7;
  const int hi   = (lane & 16) ? 8 : 0;
  const int n16  = lane & 15;

  // ---- Phase 1: QKV = x_block(64x512) @ w_slice(192x512)^T ----
  v8f acc[12];
#pragma unroll
  for (int i = 0; i < 12; ++i) acc[i] = zero8();

  for (int kk = 0; kk < DD; kk += 32) {
    // stage x tile 64x32 (f32 -> bf16), vectorized: 512 float4 chunks
    for (int e = tid; e < 512; e += 128) {
      int r = e >> 3, c4 = e & 7;
      const float* p = x + ((size_t)(b * TT + r)) * DD + kk + c4 * 4;
      const float4 f = *(const float4*)p;
      v4bf o;
      o[0] = (bf16)f.x; o[1] = (bf16)f.y; o[2] = (bf16)f.z; o[3] = (bf16)f.w;
      ((v4bf*)XA)[e] = o;
      if (kk + 32 < DD) __builtin_prefetch(p + 32, 0, 0);  // next K-slice
    }
    // stage w slice tile 192x32 (bf16 copy), vectorized: 768 uint4 chunks
    for (int e = tid; e < 768; e += 128) {
      int r = e >> 2, c8 = e & 3;
      int s = r >> 6, lr = r & 63;
      ((uint4*)WT)[e] =
          *(const uint4*)(wqkv + ((size_t)(s * DD + h * HSS + lr)) * DD + kk + c8 * 8);
    }
    __syncthreads();
    v16bf a = load_a_bf(XA, 32, wave * 16, 0, lane);
#pragma unroll
    for (int nt = 0; nt < 12; ++nt) {
      v16bf f = load_bT_bf(WT, 32, nt * 16, 0, lane);
      acc[nt] = wmma_bf(a, f, acc[nt]);
    }
    // Pin schedule: region holds exactly 26 DS reads (A=2, 12 B frags=24) + 12 WMMA.
    // Preload A + 2 B frags (6 reads), then alternate {1 WMMA, 2 reads} x10, tail 2 WMMA:
    // keeps one fragment in flight past every WMMA instead of load->wait(0)->wmma.
    SGB(0x100, 6);
#pragma unroll
    for (int it = 0; it < 10; ++it) {
      SGB(0x008, 1);
      SGB(0x100, 2);
    }
    SGB(0x008, 2);
    __syncthreads();
  }
  // scatter accumulators to Q/K/V LDS tiles (scale 1/sqrt(64) folded into q; V stored transposed)
#pragma unroll
  for (int nt = 0; nt < 12; ++nt) {
    int sel = nt >> 2;
    int n = (nt & 3) * 16 + n16;
#pragma unroll
    for (int r = 0; r < 8; ++r) {
      int m = wave * 16 + hi + r;
      float v = acc[nt][r];
      if (sel == 0)      Qs[m * 64 + n] = (bf16)(v * 0.125f);
      else if (sel == 1) Ks[m * 64 + n] = (bf16)v;
      else               Vt[n * 64 + m] = (bf16)v;   // (d, j)
    }
  }
  __syncthreads();

  // ---- Phase 2: stage rel_k (bf16, padded 240x64), vectorized ----
  for (int e = tid; e < 1920; e += 128) ((uint4*)RKs)[e] = ((const uint4*)relk)[e];
  __syncthreads();

  // ---- Phase 3: P = Q @ rel_k^T (64x240); Phase 4: S = Q@K^T + gather(P) ----
  {
    v16bf a0 = load_a_bf(Qs, 64, wave * 16, 0, lane);
    v16bf a1 = load_a_bf(Qs, 64, wave * 16, 32, lane);
    // tiles processed in pairs: 4 fragments (8 ds_load_b128) per batch
#pragma unroll
    for (int g = 0; g < 7; ++g) {
      int nt = g * 2;
      v16bf f0 = load_bT_bf(RKs, 64, nt * 16, 0, lane);
      v16bf f1 = load_bT_bf(RKs, 64, nt * 16, 32, lane);
      v16bf f2 = load_bT_bf(RKs, 64, (nt + 1) * 16, 0, lane);
      v16bf f3 = load_bT_bf(RKs, 64, (nt + 1) * 16, 32, lane);
      v8f c0 = zero8(), c1 = zero8();
      c0 = wmma_bf(a0, f0, c0);
      c0 = wmma_bf(a1, f1, c0);
      c1 = wmma_bf(a0, f2, c1);
      c1 = wmma_bf(a1, f3, c1);
#pragma unroll
      for (int r = 0; r < 8; ++r) {
        int row = (wave * 16 + hi + r) * NBP_K + n16;
        Psb[row + nt * 16]       = (bf16)c0[r];
        Psb[row + (nt + 1) * 16] = (bf16)c1[r];
      }
    }
    {  // leftover tile nt = 14
      v16bf f0 = load_bT_bf(RKs, 64, 14 * 16, 0, lane);
      v16bf f1 = load_bT_bf(RKs, 64, 14 * 16, 32, lane);
      v8f c = zero8();
      c = wmma_bf(a0, f0, c);
      c = wmma_bf(a1, f1, c);
#pragma unroll
      for (int r = 0; r < 8; ++r)
        Psb[(wave * 16 + hi + r) * NBP_K + 14 * 16 + n16] = (bf16)c[r];
    }
    __syncthreads();
    // S = Q@K^T + gathered rel bias, tiles in pairs
#pragma unroll
    for (int g = 0; g < 2; ++g) {
      int nt = g * 2;
      v16bf f0 = load_bT_bf(Ks, 64, nt * 16, 0, lane);
      v16bf f1 = load_bT_bf(Ks, 64, nt * 16, 32, lane);
      v16bf f2 = load_bT_bf(Ks, 64, (nt + 1) * 16, 0, lane);
      v16bf f3 = load_bT_bf(Ks, 64, (nt + 1) * 16, 32, lane);
      v8f c0 = zero8(), c1 = zero8();
      c0 = wmma_bf(a0, f0, c0);
      c0 = wmma_bf(a1, f1, c0);
      c1 = wmma_bf(a0, f2, c1);
      c1 = wmma_bf(a1, f3, c1);
#pragma unroll
      for (int t = 0; t < 2; ++t) {
        int j = (nt + t) * 16 + n16;
        int rj = j >> 3, fj = j & 7;
#pragma unroll
        for (int r = 0; r < 8; ++r) {
          int i = wave * 16 + hi + r;
          int ri = i >> 3, fi = i & 7;
          int bk = (rj - ri + 7) * 15 + (fj - fi + 7);   // displacement bucket, computed inline
          float cv = t == 0 ? c0[r] : c1[r];
          Ss[i * 64 + j] = cv + (float)Psb[i * NBP_K + bk];
        }
      }
    }
  }
  __syncthreads();

  // ---- Phase 5: zero W (float4), per-row softmax + bucket scatter ----
  {
    const float4 z4 = {0.f, 0.f, 0.f, 0.f};
    for (int e = tid; e < 4096; e += 128) ((float4*)Ws)[e] = z4;
  }
  __syncthreads();
  if (tid < 64) {
    const int i = tid;
    float mx = -3.0e38f;
    for (int j = 0; j < 64; ++j) mx = fmaxf(mx, Ss[i * 64 + j]);
    float sum = 0.f;
    for (int j = 0; j < 64; ++j) {
      float ev = __expf(Ss[i * 64 + j] - mx);
      Ss[i * 64 + j] = ev;
      sum += ev;
    }
    float inv = 1.f / sum;
    int ri = i >> 3, fi = i & 7;
    for (int j = 0; j < 64; ++j) {
      float a = Ss[i * 64 + j] * inv;
      As[i * 64 + j] = (bf16)a;
      int bk = ((j >> 3) - ri + 7) * 15 + ((j & 7) - fi + 7);
      Ws[i * NBP_V + bk] += a;   // row-private: no atomics needed
    }
  }
  __syncthreads();

  // ---- Phase 6: O = attn@V + W@rel_v; rel_v^T B-frags stream from global (L2-resident) ----
  {
    v16bf aA0 = load_a_bf(As, 64, wave * 16, 0, lane);
    v16bf aA1 = load_a_bf(As, 64, wave * 16, 32, lane);
    v16bf aW[8];
#pragma unroll
    for (int k8 = 0; k8 < 8; ++k8)
      aW[k8] = load_a_f32(Ws, NBP_V, wave * 16, k8 * 32, lane);
#pragma unroll
    for (int nt = 0; nt < 4; ++nt) {
      v16bf bv0 = load_bT_bf(Vt, 64, nt * 16, 0, lane);
      v16bf bv1 = load_bT_bf(Vt, 64, nt * 16, 32, lane);
      v8f c = zero8();
      c = wmma_bf(aA0, bv0, c);
      c = wmma_bf(aA1, bv1, c);
#pragma unroll
      for (int g = 0; g < 2; ++g) {         // batches of 4 global B fragments
        v16bf f0 = load_bT_bf(relvT, NBP_V, nt * 16, (g * 4 + 0) * 32, lane);
        v16bf f1 = load_bT_bf(relvT, NBP_V, nt * 16, (g * 4 + 1) * 32, lane);
        v16bf f2 = load_bT_bf(relvT, NBP_V, nt * 16, (g * 4 + 2) * 32, lane);
        v16bf f3 = load_bT_bf(relvT, NBP_V, nt * 16, (g * 4 + 3) * 32, lane);
        c = wmma_bf(aW[g * 4 + 0], f0, c);
        c = wmma_bf(aW[g * 4 + 1], f1, c);
        c = wmma_bf(aW[g * 4 + 2], f2, c);
        c = wmma_bf(aW[g * 4 + 3], f3, c);
      }
      int d0 = nt * 16 + n16;
#pragma unroll
      for (int r = 0; r < 8; ++r) {
        int i = wave * 16 + hi + r;
        attn_out[((size_t)(b * TT + i)) * DD + h * HSS + d0] = (bf16)c[r];
      }
    }
  }
}

// ---------------- kernel 2: output projection GEMM (65536x512x512) + bias ----------------
__global__ __launch_bounds__(128)
void proj_gemm(const bf16* __restrict__ ain, const bf16* __restrict__ wp,
               const float* __restrict__ bias, float* __restrict__ out) {
  __shared__ __align__(16) bf16 At[64 * 32];
  __shared__ __align__(16) bf16 Bt[64 * 32];
  const int tid  = threadIdx.x;
  const int wave = tid >> 5;
  const int lane = tid & 31;
  const int m0 = blockIdx.x * 64;
  const int n0 = blockIdx.y * 64;

  v8f acc[4];
#pragma unroll
  for (int i = 0; i < 4; ++i) acc[i] = zero8();

  for (int kk = 0; kk < DD; kk += 32) {
    for (int e = tid; e < 256; e += 128) {  // 256 uint4 chunks (8 bf16 each)
      int r = e >> 2, c8 = e & 3;
      ((uint4*)At)[e] = *(const uint4*)(ain + (size_t)(m0 + r) * DD + kk + c8 * 8);
    }
    for (int e = tid; e < 256; e += 128) {
      int r = e >> 2, c8 = e & 3;
      ((uint4*)Bt)[e] = *(const uint4*)(wp + (size_t)(n0 + r) * DD + kk + c8 * 8);
    }
    __syncthreads();
    v16bf a = load_a_bf(At, 32, wave * 16, 0, lane);
#pragma unroll
    for (int nt = 0; nt < 4; ++nt) {
      v16bf f = load_bT_bf(Bt, 32, nt * 16, 0, lane);
      acc[nt] = wmma_bf(a, f, acc[nt]);
    }
    // Region: 10 DS reads (A=2, 4 B frags=8) + 4 WMMA. Preload 4 reads, alternate, tail WMMA.
    SGB(0x100, 4);
#pragma unroll
    for (int it = 0; it < 3; ++it) {
      SGB(0x008, 1);
      SGB(0x100, 2);
    }
    SGB(0x008, 1);
    __syncthreads();
  }
  const int hi  = (lane & 16) ? 8 : 0;
  const int n16 = lane & 15;
#pragma unroll
  for (int nt = 0; nt < 4; ++nt) {
    int n = n0 + nt * 16 + n16;
    float bv = bias[n];
#pragma unroll
    for (int r = 0; r < 8; ++r) {
      int m = m0 + wave * 16 + hi + r;
      out[(size_t)m * DD + n] = acc[nt][r] + bv;
    }
  }
}

// ---------------- launch ----------------
extern "C" void kernel_launch(void* const* d_in, const int* in_sizes, int n_in,
                              void* d_out, int out_size, void* d_ws, size_t ws_size,
                              hipStream_t stream) {
  const float* x      = (const float*)d_in[0];
  const float* w_qkv  = (const float*)d_in[1];
  const float* w_proj = (const float*)d_in[2];
  const float* b_proj = (const float*)d_in[3];
  const float* rel_k  = (const float*)d_in[4];
  const float* rel_v  = (const float*)d_in[5];

  char* ws = (char*)d_ws;
  bf16* wqkvb  = (bf16*)(ws);                          // 1,572,864 B
  bf16* wprojb = (bf16*)(ws + 1572864);                //   524,288 B
  bf16* relkp  = (bf16*)(ws + 2097152);                //    30,720 B
  bf16* relvT  = (bf16*)(ws + 2127872);                //    32,768 B (transposed (d,bucket))
  bf16* aout   = (bf16*)(ws + 2160640);                // 67,108,864 B
  float* out   = (float*)d_out;

  convert_kernel<<<N_CONV / 256, 256, 0, stream>>>(w_qkv, w_proj, rel_k, rel_v,
                                                   wqkvb, wprojb, relkp, relvT);
  attn_fused<<<BB * NHH, 128, SMEM_BYTES, stream>>>(x, wqkvb, relkp, relvT, aout);
  proj_gemm<<<dim3((BB * TT) / 64, DD / 64), 128, 0, stream>>>(aout, wprojb, b_proj, out);
}
